// Transformer_ours_91147795956443
// MI455X (gfx1250) — matrix operand order repaired and checked by reading, not compiled, not run on hardware
//
#include <hip/hip_runtime.h>
#include <hip/hip_fp16.h>

// ---------------------------------------------------------------------------
// MI455X (gfx1250) implementation of the compression-scan transformer.
// All GEMMs go through v_wmma_f32_16x16x32_f16 (f16 inputs, f32 accum).
//  - Weight GEMMs: weights pre-packed to WMMA-fragment-major f16 (one 32B
//    contiguous load per lane-fragment); 32x64 output per wave (2x B reuse).
//  - Attention GEMMs: B operand always contiguous along K (K^T naturally,
//    V via a cheap per-head transpose), loaded as b128 + cvt_pk_f16_f32.
//  - global_prefetch_b8 pulls the next K panel while WMMAs retire.
// ---------------------------------------------------------------------------

typedef __attribute__((ext_vector_type(16))) _Float16 v16h;
typedef __attribute__((ext_vector_type(8)))  float    v8f;

#define B_    16
#define LENC_ 2048
#define MCH   4
#define LSEQ  512          // L = LENC/M
#define FIN   128
#define FH    64           // F_IN/2
#define CATW  192          // 3*F_IN/2
#define DIM   512
#define NHEAD 8
#define HD_   64
#define DFF_  2048
#define LDE   128
#define CK_   10

// ---------------------------------------------------------------------------
// Weight packing: W[K,N] f32 row-major -> fragment-major f16.
// Fragment (kT, nT) holds a 32x16 B tile; element (lane, j) = W[k, n] with
//   n = nT*16 + (lane & 15),  k = kT*32 + (lane < 16 ? 0 : 8) + (j < 8 ? j : j + 8)
// stored at P[(kT*(N/16) + nT)*512 + lane*16 + j].
// ---------------------------------------------------------------------------
__global__ void pack_wgt(const float* __restrict__ W, _Float16* __restrict__ P,
                         int N, int K)
{
    const size_t idx = (size_t)blockIdx.x * blockDim.x + threadIdx.x; // K*N
    const int lanej = (int)(idx & 511);
    const int frag  = (int)(idx >> 9);
    const int nT = frag % (N >> 4);
    const int kT = frag / (N >> 4);
    const int lane = lanej >> 4;
    const int j    = lanej & 15;
    const int n = nT * 16 + (lane & 15);
    const int k = kT * 32 + (lane < 16 ? 0 : 8) + (j < 8 ? j : j + 8);
    P[idx] = (_Float16)W[(size_t)k * N + n];
}

// ---------------------------------------------------------------------------
// Per-head transpose: VT[z, d, s] = V[zb*sV1 + s*ldv + zh*64 + d]
// (z = zb*nH + zh). Makes the P@V B operand contiguous along K.
// ---------------------------------------------------------------------------
__global__ void transpose_heads(const float* __restrict__ V, float* __restrict__ VT,
                                int Lk, int ldv, long long sV1, int nH)
{
    const long long idx = (long long)blockIdx.x * blockDim.x + threadIdx.x; // nz*HD*Lk
    const int s = (int)(idx % Lk);
    const long long zd = idx / Lk;
    const int d = (int)(zd % HD_);
    const int z = (int)(zd / HD_);
    const int zb = z / nH, zh = z % nH;
    VT[idx] = V[(long long)zb * sV1 + (long long)s * ldv + zh * HD_ + d];
}

// ---------------------------------------------------------------------------
// Packed-weight WMMA GEMM: C = A @ Wpk + bias [, relu]
// Each wave computes 32 rows x 64 cols (8 WMMAs per K-step, B reused 2x).
// Requires: M % 16 == 0 (M tile 256/block), N % 64 == 0, K % 32 == 0.
// ---------------------------------------------------------------------------
__global__ __launch_bounds__(256)
void wmma_gemm_pk_k(const float* __restrict__ A, const _Float16* __restrict__ Bpk,
                    const float* __restrict__ bias, float* __restrict__ C,
                    int M, int N, int K, int lda, int ldc, int relu)
{
    const int wave = threadIdx.x >> 5;
    const int lane = threadIdx.x & 31;
    const int r    = lane & 15;
    const bool hi  = lane >= 16;

    const int row0 = ((int)blockIdx.y * 8 + wave) * 32;
    const int col0 = (int)blockIdx.x * 64;
    if (row0 >= M) return;                    // wave-uniform
    const int nT16 = N >> 4;

    v8f acc[2][4] = {};
    const float* arp0 = A + (long long)(row0 + r) * lda;
    const float* arp1 = A + (long long)(row0 + 16 + r) * lda;

    for (int k0 = 0; k0 < K; k0 += 32) {
        const float* ap0 = arp0 + k0 + (hi ? 8 : 0);
        const float* ap1 = arp1 + k0 + (hi ? 8 : 0);
        const _Float16* bbase = Bpk +
            (((size_t)(k0 >> 5) * nT16 + (col0 >> 4)) << 9) + (lane << 4);
        if (k0 + 32 < K) {
            __builtin_prefetch((const void*)(ap0 + 32), 0, 1);
            __builtin_prefetch((const void*)(ap1 + 32), 0, 1);
            __builtin_prefetch((const void*)(bbase + ((size_t)nT16 << 9)), 0, 1);
        }
        v16h a0, a1;
        #pragma unroll
        for (int j = 0; j < 8; ++j) {
            a0[j]     = (_Float16)ap0[j];
            a0[8 + j] = (_Float16)ap0[16 + j];
            a1[j]     = (_Float16)ap1[j];
            a1[8 + j] = (_Float16)ap1[16 + j];
        }
        #pragma unroll
        for (int t = 0; t < 4; ++t) {
            const v16h b = *reinterpret_cast<const v16h*>(bbase + ((size_t)t << 9));
            acc[0][t] = __builtin_amdgcn_wmma_f32_16x16x32_f16(
                            false, a0, false, b, (short)0, acc[0][t], false, false);
            acc[1][t] = __builtin_amdgcn_wmma_f32_16x16x32_f16(
                            false, a1, false, b, (short)0, acc[1][t], false, false);
        }
    }

    #pragma unroll
    for (int u = 0; u < 2; ++u) {
        #pragma unroll
        for (int t = 0; t < 4; ++t) {
            const int col = col0 + t * 16 + r;
            const float bv = bias ? bias[col] : 0.0f;
            #pragma unroll
            for (int p = 0; p < 8; ++p) {
                const int row = row0 + u * 16 + p + (hi ? 8 : 0);
                if (row < M) {
                    float v = acc[u][t] [p] + bv;
                    if (relu) v = fmaxf(v, 0.0f);
                    C[(long long)row * ldc + col] = v;
                }
            }
        }
    }
}

// ---------------------------------------------------------------------------
// Batched attention WMMA GEMM with B contiguous along K:
//   C = alpha * (A @ B), B(k,n) = Bt[n*ldb + k]
// Each wave computes 16 rows x 64 cols. No bias/relu (attention only).
// ---------------------------------------------------------------------------
__global__ __launch_bounds__(256)
void wmma_gemm_tn_k(const float* __restrict__ A, const float* __restrict__ Bt,
                    float* __restrict__ C,
                    int M, int N, int K, int lda, int ldb, int ldc,
                    int nH, long long sA1, long long sA2,
                    long long sB1, long long sB2,
                    long long sC1, long long sC2, float alpha)
{
    const int wave = threadIdx.x >> 5;
    const int lane = threadIdx.x & 31;
    const int r    = lane & 15;
    const bool hi  = lane >= 16;

    const int row0 = ((int)blockIdx.y * 8 + wave) * 16;
    const int col0 = (int)blockIdx.x * 64;
    if (row0 >= M) return;                    // wave-uniform

    const int zb = (int)blockIdx.z / nH;
    const int zh = (int)blockIdx.z % nH;
    const float* Ab = A  + (long long)zb * sA1 + (long long)zh * sA2;
    const float* Bb = Bt + (long long)zb * sB1 + (long long)zh * sB2;
    float*       Cb = C  + (long long)zb * sC1 + (long long)zh * sC2;

    v8f acc[4] = {};
    const float* arp = Ab + (long long)(row0 + r) * lda;

    for (int k0 = 0; k0 < K; k0 += 32) {
        const float* ap = arp + k0 + (hi ? 8 : 0);
        if (k0 + 32 < K) __builtin_prefetch((const void*)(ap + 32), 0, 1);
        v16h a;
        #pragma unroll
        for (int j = 0; j < 8; ++j) {
            a[j]     = (_Float16)ap[j];
            a[8 + j] = (_Float16)ap[16 + j];
        }
        const int kb = k0 + (hi ? 8 : 0);
        #pragma unroll
        for (int t = 0; t < 4; ++t) {
            const float* bp = Bb + (long long)(col0 + t * 16 + r) * ldb + kb;
            v16h b;
            #pragma unroll
            for (int j = 0; j < 16; ++j) b[j] = (_Float16)bp[(j & 7) + ((j >> 3) << 4)];
            acc[t] = __builtin_amdgcn_wmma_f32_16x16x32_f16(
                         false, a, false, b, (short)0, acc[t], false, false);
        }
    }

    #pragma unroll
    for (int t = 0; t < 4; ++t) {
        const int col = col0 + t * 16 + r;
        #pragma unroll
        for (int p = 0; p < 8; ++p) {
            const int row = row0 + p + (hi ? 8 : 0);
            if (row < M) Cb[(long long)row * ldc + col] = acc[t][p] * alpha;
        }
    }
}

// ---------------------------------------------------------------------------
// Softmax over rows of length N (N in {128, 512}); one wave per row.
// ---------------------------------------------------------------------------
template <int N>
__global__ __launch_bounds__(256)
void softmax_rows(float* __restrict__ S, int rows)
{
    const int wave = threadIdx.x >> 5, lane = threadIdx.x & 31;
    const int row = (int)blockIdx.x * 8 + wave;
    if (row >= rows) return;
    float* p = S + (long long)row * N;
    constexpr int C = N / 32;
    float v[C];
    float mx = -3.0e38f;
    #pragma unroll
    for (int i = 0; i < C; ++i) { v[i] = p[i * 32 + lane]; mx = fmaxf(mx, v[i]); }
    #pragma unroll
    for (int off = 16; off; off >>= 1) mx = fmaxf(mx, __shfl_xor(mx, off));
    float s = 0.0f;
    #pragma unroll
    for (int i = 0; i < C; ++i) { v[i] = __expf(v[i] - mx); s += v[i]; }
    #pragma unroll
    for (int off = 16; off; off >>= 1) s += __shfl_xor(s, off);
    const float inv = 1.0f / s;
    #pragma unroll
    for (int i = 0; i < C; ++i) p[i * 32 + lane] = v[i] * inv;
}

// ---------------------------------------------------------------------------
// out[row,:] = LN(X[row,:] + R[row,:]; g,b), rows of width DIM=512.
// ---------------------------------------------------------------------------
__global__ __launch_bounds__(256)
void ln_rows(const float* __restrict__ X, const float* __restrict__ R,
             const float* __restrict__ g, const float* __restrict__ b,
             float* __restrict__ O, int rows, int ldo)
{
    const int wave = threadIdx.x >> 5, lane = threadIdx.x & 31;
    const int row = (int)blockIdx.x * 8 + wave;
    if (row >= rows) return;
    const float* xp = X + (long long)row * DIM;
    const float* rp = R + (long long)row * DIM;
    float v[16];
    float s = 0.0f, sq = 0.0f;
    #pragma unroll
    for (int i = 0; i < 16; ++i) {
        float t = xp[i * 32 + lane] + rp[i * 32 + lane];
        v[i] = t; s += t; sq += t * t;
    }
    #pragma unroll
    for (int off = 16; off; off >>= 1) { s += __shfl_xor(s, off); sq += __shfl_xor(sq, off); }
    const float mu  = s * (1.0f / DIM);
    const float var = sq * (1.0f / DIM) - mu * mu;     // biased var (jnp.var)
    const float rs  = rsqrtf(var + 1e-5f);
    float* op = O + (long long)row * ldo;
    #pragma unroll
    for (int i = 0; i < 16; ++i) {
        const int c = i * 32 + lane;
        op[c] = g[c] * (v[i] - mu) * rs + b[c];
    }
}

// ---------------------------------------------------------------------------
// Causal scalar-tap conv along seq: out[b,s,d] = sum_i k[i]*x[b,s+i-9,d]
// ---------------------------------------------------------------------------
__global__ void causal_conv(const float* __restrict__ x, const float* __restrict__ kw,
                            float* __restrict__ out, int ldo)
{
    long long idx = (long long)blockIdx.x * blockDim.x + threadIdx.x;  // B*L*D
    const int d = (int)(idx % DIM);
    const long long bs = idx / DIM;
    const int s = (int)(bs % LSEQ);
    const int b = (int)(bs / LSEQ);
    float acc = 0.0f;
    #pragma unroll
    for (int i = 0; i < CK_; ++i) {
        const int sp = s + i - (CK_ - 1);
        if (sp >= 0) acc += kw[i] * x[((long long)b * LSEQ + sp) * DIM + d];
    }
    out[((long long)b * LSEQ + s) * ldo + d] = acc;
}

// ---------------------------------------------------------------------------
// Compression-scan step: h_out[row,col] = bcomp[col] + [chunk|h_in](row,:) @ Wcomp[:,col]
// ---------------------------------------------------------------------------
__global__ void comp_step(const float* __restrict__ chunk, const float* __restrict__ hin,
                          const float* __restrict__ Wc, const float* __restrict__ bc,
                          float* __restrict__ hout)
{
    const int idx = (int)blockIdx.x * blockDim.x + threadIdx.x;   // LSEQ*FH
    const int col = idx & (FH - 1);
    const int row = idx >> 6;
    float acc = bc[col];
    const float* cr = chunk + (long long)row * FIN;
    #pragma unroll 4
    for (int k = 0; k < FIN; ++k) acc += cr[k] * Wc[k * FH + col];
    const float* hr = hin + (long long)row * FH;
    #pragma unroll 4
    for (int k = 0; k < FH; ++k) acc += hr[k] * Wc[(FIN + k) * FH + col];
    hout[(long long)row * FH + col] = acc;
}

__global__ void cat_copy(const float* __restrict__ chunk, const float* __restrict__ hin,
                         float* __restrict__ dst)
{
    const int idx = (int)blockIdx.x * blockDim.x + threadIdx.x;   // LSEQ*CATW
    const int col = idx % CATW;
    const int row = idx / CATW;
    dst[idx] = (col < FIN) ? chunk[(long long)row * FIN + col]
                           : hin[(long long)row * FH + (col - FIN)];
}

__global__ void zero_f32(float* __restrict__ p, int n)
{
    const int i = (int)blockIdx.x * blockDim.x + threadIdx.x;
    if (i < n) p[i] = 0.0f;
}

// ---------------------------------------------------------------------------
// out[row] = T[row,:] . Wmap + bmap ; one wave per row.
// ---------------------------------------------------------------------------
__global__ __launch_bounds__(256)
void map_out_kernel(const float* __restrict__ T, const float* __restrict__ Wm,
                    const float* __restrict__ bm, float* __restrict__ out, int rows)
{
    const int wave = threadIdx.x >> 5, lane = threadIdx.x & 31;
    const int row = (int)blockIdx.x * 8 + wave;
    if (row >= rows) return;
    const float* tp = T + (long long)row * DIM;
    float s = 0.0f;
    #pragma unroll
    for (int i = 0; i < 16; ++i) s += tp[i * 32 + lane] * Wm[i * 32 + lane];
    #pragma unroll
    for (int off = 16; off; off >>= 1) s += __shfl_xor(s, off);
    if (lane == 0) out[row] = s + bm[0];
}

// ---------------------------------------------------------------------------
// Host-side helpers
// ---------------------------------------------------------------------------
static inline void gemm_pk(hipStream_t st, const float* A, const _Float16* Bpk,
                           const float* bias, float* C,
                           int M, int N, int K, int lda, int ldc, int relu = 0)
{
    dim3 grid(N / 64, (M + 255) / 256, 1);
    wmma_gemm_pk_k<<<grid, dim3(256), 0, st>>>(A, Bpk, bias, C, M, N, K, lda, ldc, relu);
}

static inline void gemm_tn(hipStream_t st, const float* A, const float* Bt, float* C,
                           int M, int N, int K, int lda, int ldb, int ldc,
                           int nz, int nH,
                           long long sA1, long long sA2,
                           long long sB1, long long sB2,
                           long long sC1, long long sC2, float alpha)
{
    dim3 grid(N / 64, (M + 127) / 128, nz);
    wmma_gemm_tn_k<<<grid, dim3(256), 0, st>>>(A, Bt, C, M, N, K, lda, ldb, ldc,
                                               nH, sA1, sA2, sB1, sB2, sC1, sC2, alpha);
}

extern "C" void kernel_launch(void* const* d_in, const int* in_sizes, int n_in,
                              void* d_out, int out_size, void* d_ws, size_t ws_size,
                              hipStream_t stream)
{
    (void)in_sizes; (void)n_in; (void)out_size; (void)ws_size;

    const float* src     = (const float*)d_in[0];
    const float* tgt     = (const float*)d_in[1];
    const float* Wcomp   = (const float*)d_in[2];
    const float* bcomp   = (const float*)d_in[3];
    const float* Wenc_in = (const float*)d_in[4];
    const float* benc_in = (const float*)d_in[5];
    const float* Wdec_in = (const float*)d_in[6];
    const float* bdec_in = (const float*)d_in[7];
    const float* Wmap    = (const float*)d_in[8];
    const float* bmap    = (const float*)d_in[9];
    const float* eWqkv = (const float*)d_in[10];
    const float* ebqkv = (const float*)d_in[11];
    const float* eWo   = (const float*)d_in[12];
    const float* ebo   = (const float*)d_in[13];
    const float* eln1g = (const float*)d_in[14];
    const float* eln1b = (const float*)d_in[15];
    const float* eln2g = (const float*)d_in[16];
    const float* eln2b = (const float*)d_in[17];
    const float* eW1   = (const float*)d_in[18];
    const float* eb1   = (const float*)d_in[19];
    const float* eW2   = (const float*)d_in[20];
    const float* eb2   = (const float*)d_in[21];
    const float* ekconv= (const float*)d_in[22];
    const float* eWcmb = (const float*)d_in[23];
    const float* ebcmb = (const float*)d_in[24];
    const float* dWqkv = (const float*)d_in[25];
    const float* dbqkv = (const float*)d_in[26];
    const float* dWo_s = (const float*)d_in[27];
    const float* dbo_s = (const float*)d_in[28];
    const float* dWq   = (const float*)d_in[29];
    const float* dbq   = (const float*)d_in[30];
    const float* dWkv  = (const float*)d_in[31];
    const float* dbkv  = (const float*)d_in[32];
    const float* dWo_c = (const float*)d_in[33];
    const float* dbo_c = (const float*)d_in[34];
    const float* dln1g = (const float*)d_in[35];
    const float* dln1b = (const float*)d_in[36];
    const float* dln2g = (const float*)d_in[37];
    const float* dln2b = (const float*)d_in[38];
    const float* dln3g = (const float*)d_in[39];
    const float* dln3b = (const float*)d_in[40];
    const float* dW1   = (const float*)d_in[41];
    const float* db1   = (const float*)d_in[42];
    const float* dW2   = (const float*)d_in[43];
    const float* db2   = (const float*)d_in[44];

    float* ws = (float*)d_ws;

    // ---- workspace layout (floats) ----------------------------------------
    const size_t OH0  = 0;
    const size_t OH1  = OH0  + (size_t)LSEQ * FH;                 // 32768
    const size_t ONS  = OH1  + (size_t)LSEQ * FH;
    const size_t OX   = ONS  + (size_t)B_ * LSEQ * CATW;          // new_src
    const size_t OQKV = OX   + (size_t)B_ * LSEQ * DIM;           // x / mem
    const size_t OBIG = OQKV + (size_t)B_ * LSEQ * 3 * DIM;       // qkv / temps
    const size_t OCAT = OBIG + (size_t)B_ * NHEAD * LSEQ * LSEQ;  // scores / ffn
    const size_t OCTX = OCAT + (size_t)B_ * LSEQ * 2 * DIM;       // concat / V^T
    const size_t OX1  = OCTX + (size_t)B_ * LSEQ * DIM;           // attn ctx
    const size_t OPK  = OX1  + (size_t)B_ * LSEQ * DIM;           // packed f16 weights

    float* h0   = ws + OH0;
    float* h1   = ws + OH1;
    float* nsrc = ws + ONS;
    float* x    = ws + OX;     // encoder activations, then mem
    float* qkv  = ws + OQKV;   // 12.58M floats
    float* big  = ws + OBIG;   // 33.55M floats (scores / FFN hidden)
    float* cat  = ws + OCAT;   // concat buffer, doubles as V^T scratch
    float* ctx  = ws + OCTX;
    float* x1   = ws + OX1;
    float* vt   = cat;         // V^T lives in cat region while attention runs

    // decoder sub-buffers (reuse large regions)
    float* tbuf = x1;                               // [2048, 512]
    float* t1   = x1 + (size_t)B_ * LDE * DIM;      // [2048, 512]
    float* t2   = t1 + (size_t)B_ * LDE * DIM;      // [2048, 512]
    float* dq   = qkv + (size_t)B_ * LSEQ * 2 * DIM;        // after dkv region
    float* tmp  = dq  + (size_t)B_ * LDE * DIM;             // 1M temp

    const int NROW_E = B_ * LSEQ;   // 8192
    const int NROW_D = B_ * LDE;    // 2048
    const int NZ = B_ * NHEAD;      // 128
    const float ATT_SC = 0.125f;    // 1/sqrt(64)

    // ------------------------------------------------------------------
    // 0) pack all weight matrices to fragment-major f16 (hits L2 afterwards)
    // ------------------------------------------------------------------
    _Float16* q = (_Float16*)(ws + OPK);
    auto packW = [&](const float* W, int K, int N) -> _Float16* {
        _Float16* dst = q;
        pack_wgt<<<(unsigned)(((size_t)K * N) / 256), 256, 0, stream>>>(W, dst, N, K);
        q += (size_t)K * N;
        return dst;
    };
    _Float16* pk_enc_in = packW(Wenc_in, CATW, DIM);
    _Float16* pk_dec_in = packW(Wdec_in, FIN, DIM);
    _Float16 *pk_eqkv[2], *pk_eWo[2], *pk_eW1[2], *pk_eW2[2], *pk_ecmb[2];
    _Float16 *pk_dqkv[2], *pk_dWos[2], *pk_dWq[2], *pk_dkvw[2], *pk_dWoc[2], *pk_dW1[2], *pk_dW2[2];
    for (int l = 0; l < 2; ++l) {
        pk_eqkv[l] = packW(eWqkv + (size_t)l * DIM * 3 * DIM, DIM, 3 * DIM);
        pk_eWo[l]  = packW(eWo   + (size_t)l * DIM * DIM,     DIM, DIM);
        pk_eW1[l]  = packW(eW1   + (size_t)l * DIM * DFF_,    DIM, DFF_);
        pk_eW2[l]  = packW(eW2   + (size_t)l * DFF_ * DIM,    DFF_, DIM);
        pk_ecmb[l] = packW(eWcmb + (size_t)l * 2 * DIM * DIM, 2 * DIM, DIM);
        pk_dqkv[l] = packW(dWqkv + (size_t)l * DIM * 3 * DIM, DIM, 3 * DIM);
        pk_dWos[l] = packW(dWo_s + (size_t)l * DIM * DIM,     DIM, DIM);
        pk_dWq[l]  = packW(dWq   + (size_t)l * DIM * DIM,     DIM, DIM);
        pk_dkvw[l] = packW(dWkv  + (size_t)l * DIM * 2 * DIM, DIM, 2 * DIM);
        pk_dWoc[l] = packW(dWo_c + (size_t)l * DIM * DIM,     DIM, DIM);
        pk_dW1[l]  = packW(dW1   + (size_t)l * DIM * DFF_,    DIM, DFF_);
        pk_dW2[l]  = packW(dW2   + (size_t)l * DFF_ * DIM,    DFF_, DIM);
    }

    // ------------------------------------------------------------------
    // 1) sequential long-dependency compression scan (64 chunks)
    // ------------------------------------------------------------------
    zero_f32<<<(LSEQ * FH + 255) / 256, 256, 0, stream>>>(h0, LSEQ * FH);
    float* hb[2] = { h0, h1 };
    for (int i = 0; i < B_ * MCH; ++i) {
        const float* chunk = src + (size_t)i * LSEQ * FIN;
        float* hin  = hb[i & 1];
        float* hout = hb[(i + 1) & 1];
        if ((i & 3) == 3)
            cat_copy<<<(LSEQ * CATW) / 256, 256, 0, stream>>>(
                chunk, hin, nsrc + (size_t)(i >> 2) * LSEQ * CATW);
        comp_step<<<(LSEQ * FH) / 256, 256, 0, stream>>>(chunk, hin, Wcomp, bcomp, hout);
    }

    // x = new_src @ Wenc_in + benc_in      [8192, 512], K=192
    gemm_pk(stream, nsrc, pk_enc_in, benc_in, x, NROW_E, DIM, CATW, CATW, DIM);

    // ------------------------------------------------------------------
    // 2) encoder layers
    // ------------------------------------------------------------------
    for (int l = 0; l < 2; ++l) {
        // qkv = x @ eWqkv[l] + ebqkv[l]
        gemm_pk(stream, x, pk_eqkv[l], ebqkv + l * 3 * DIM, qkv,
                NROW_E, 3 * DIM, DIM, DIM, 3 * DIM);
        // scores = Q @ K^T / 8           per (b,h): [512,512], K=64
        gemm_tn(stream, qkv, qkv + DIM, big,
                LSEQ, LSEQ, HD_, 3 * DIM, 3 * DIM, LSEQ,
                NZ, NHEAD,
                (long long)LSEQ * 3 * DIM, HD_,
                (long long)LSEQ * 3 * DIM, HD_,
                (long long)NHEAD * LSEQ * LSEQ, (long long)LSEQ * LSEQ, ATT_SC);
        softmax_rows<LSEQ><<<(B_ * NHEAD * LSEQ) / 8, 256, 0, stream>>>(big, B_ * NHEAD * LSEQ);
        // V^T per head, then ctx = P @ V    per (b,h): [512,64], K=512
        transpose_heads<<<(unsigned)(((size_t)NZ * HD_ * LSEQ) / 256), 256, 0, stream>>>(
            qkv + 2 * DIM, vt, LSEQ, 3 * DIM, (long long)LSEQ * 3 * DIM, NHEAD);
        gemm_tn(stream, big, vt, ctx,
                LSEQ, HD_, LSEQ, LSEQ, LSEQ, DIM,
                NZ, NHEAD,
                (long long)NHEAD * LSEQ * LSEQ, (long long)LSEQ * LSEQ,
                (long long)NHEAD * HD_ * LSEQ, (long long)HD_ * LSEQ,
                (long long)LSEQ * DIM, HD_, 1.0f);
        // attn_out = ctx @ eWo + ebo  (into qkv region as temp)
        gemm_pk(stream, ctx, pk_eWo[l], ebo + l * DIM, qkv,
                NROW_E, DIM, DIM, DIM, DIM);
        // x1 = LN(x + attn_out)
        ln_rows<<<NROW_E / 8, 256, 0, stream>>>(qkv, x, eln1g + l * DIM, eln1b + l * DIM,
                                                x1, NROW_E, DIM);
        // f = relu(x1 @ eW1 + eb1) @ eW2 + eb2
        gemm_pk(stream, x1, pk_eW1[l], eb1 + l * DFF_, big,
                NROW_E, DFF_, DIM, DIM, DFF_, 1);
        gemm_pk(stream, big, pk_eW2[l], eb2 + l * DIM, qkv,
                NROW_E, DIM, DFF_, DFF_, DIM);
        // y = LN(x1 + f) -> cat[:, :512]
        ln_rows<<<NROW_E / 8, 256, 0, stream>>>(qkv, x1, eln2g + l * DIM, eln2b + l * DIM,
                                                cat, NROW_E, 2 * DIM);
        // conv(x) -> cat[:, 512:]
        causal_conv<<<(unsigned)(((size_t)NROW_E * DIM) / 256), 256, 0, stream>>>(
            x, ekconv + l * CK_, cat + DIM, 2 * DIM);
        // x = cat @ eWcmb + ebcmb
        gemm_pk(stream, cat, pk_ecmb[l], ebcmb + l * DIM, x,
                NROW_E, DIM, 2 * DIM, 2 * DIM, DIM);
    }
    // mem == x

    // ------------------------------------------------------------------
    // 3) decoder
    // ------------------------------------------------------------------
    gemm_pk(stream, tgt, pk_dec_in, bdec_in, tbuf, NROW_D, DIM, FIN, FIN, DIM);

    for (int l = 0; l < 2; ++l) {
        // ---- self attention ----
        gemm_pk(stream, tbuf, pk_dqkv[l], dbqkv + l * 3 * DIM, qkv,
                NROW_D, 3 * DIM, DIM, DIM, 3 * DIM);
        gemm_tn(stream, qkv, qkv + DIM, big,
                LDE, LDE, HD_, 3 * DIM, 3 * DIM, LDE,
                NZ, NHEAD,
                (long long)LDE * 3 * DIM, HD_,
                (long long)LDE * 3 * DIM, HD_,
                (long long)NHEAD * LDE * LDE, (long long)LDE * LDE, ATT_SC);
        softmax_rows<LDE><<<(B_ * NHEAD * LDE) / 8, 256, 0, stream>>>(big, B_ * NHEAD * LDE);
        transpose_heads<<<(unsigned)(((size_t)NZ * HD_ * LDE) / 256), 256, 0, stream>>>(
            qkv + 2 * DIM, vt, LDE, 3 * DIM, (long long)LDE * 3 * DIM, NHEAD);
        gemm_tn(stream, big, vt, ctx,
                LDE, HD_, LDE, LDE, LDE, DIM,
                NZ, NHEAD,
                (long long)NHEAD * LDE * LDE, (long long)LDE * LDE,
                (long long)NHEAD * HD_ * LDE, (long long)HD_ * LDE,
                (long long)LDE * DIM, HD_, 1.0f);
        gemm_pk(stream, ctx, pk_dWos[l], dbo_s + l * DIM, tmp,
                NROW_D, DIM, DIM, DIM, DIM);
        ln_rows<<<NROW_D / 8, 256, 0, stream>>>(tmp, tbuf, dln1g + l * DIM, dln1b + l * DIM,
                                                t1, NROW_D, DIM);

        // ---- cross attention ----
        // dkv = mem @ dWkv + dbkv   [8192, 1024]
        gemm_pk(stream, x, pk_dkvw[l], dbkv + l * 2 * DIM, qkv,
                NROW_E, 2 * DIM, DIM, DIM, 2 * DIM);
        // q = t1 @ dWq + dbq
        gemm_pk(stream, t1, pk_dWq[l], dbq + l * DIM, dq,
                NROW_D, DIM, DIM, DIM, DIM);
        // scores = Q @ Kc^T / 8      per (b,h): [128, 512], K=64
        gemm_tn(stream, dq, qkv, big,
                LDE, LSEQ, HD_, DIM, 2 * DIM, LSEQ,
                NZ, NHEAD,
                (long long)LDE * DIM, HD_,
                (long long)LSEQ * 2 * DIM, HD_,
                (long long)NHEAD * LDE * LSEQ, (long long)LDE * LSEQ, ATT_SC);
        softmax_rows<LSEQ><<<(B_ * NHEAD * LDE) / 8, 256, 0, stream>>>(big, B_ * NHEAD * LDE);
        // cctx = P @ Vc              per (b,h): [128, 64], K=512
        transpose_heads<<<(unsigned)(((size_t)NZ * HD_ * LSEQ) / 256), 256, 0, stream>>>(
            qkv + DIM, vt, LSEQ, 2 * DIM, (long long)LSEQ * 2 * DIM, NHEAD);
        gemm_tn(stream, big, vt, ctx,
                LDE, HD_, LSEQ, LSEQ, LSEQ, DIM,
                NZ, NHEAD,
                (long long)NHEAD * LDE * LSEQ, (long long)LDE * LSEQ,
                (long long)NHEAD * HD_ * LSEQ, (long long)HD_ * LSEQ,
                (long long)LDE * DIM, HD_, 1.0f);
        gemm_pk(stream, ctx, pk_dWoc[l], dbo_c + l * DIM, tmp,
                NROW_D, DIM, DIM, DIM, DIM);
        ln_rows<<<NROW_D / 8, 256, 0, stream>>>(tmp, t1, dln2g + l * DIM, dln2b + l * DIM,
                                                t2, NROW_D, DIM);

        // ---- FFN ----
        gemm_pk(stream, t2, pk_dW1[l], db1 + l * DFF_, big,
                NROW_D, DFF_, DIM, DIM, DFF_, 1);
        gemm_pk(stream, big, pk_dW2[l], db2 + l * DIM, tmp,
                NROW_D, DIM, DFF_, DFF_, DIM);
        ln_rows<<<NROW_D / 8, 256, 0, stream>>>(tmp, t2, dln3g + l * DIM, dln3b + l * DIM,
                                                tbuf, NROW_D, DIM);
    }

    // ------------------------------------------------------------------
    // 4) out[b, s] = t @ Wmap + bmap
    // ------------------------------------------------------------------
    map_out_kernel<<<NROW_D / 8, 256, 0, stream>>>(tbuf, Wmap, bmap, (float*)d_out, NROW_D);
}
